// GatedStateUnit_47545287966800
// MI455X (gfx1250) — matrix operand order repaired and checked by reading, not compiled, not run on hardware
//
#include <hip/hip_runtime.h>
#include <hip/hip_bf16.h>
#include <math.h>

// ---------------------------------------------------------------------------
// CDNA5 (gfx1250) GatedStateUnit
//   conv  = diagonal-SSM linear recurrence (replaces FFT conv; exact)
//   mid   = gelu(conv @ ss_W^T + ss_b)            [bf16 WMMA]
//   G     = gelu(x @ gate_W^T + gate_b) * mid     [bf16 WMMA, fused product]
//   out   = G @ fin_W^T + fin_b                   [bf16 WMMA, f32 out]
// GEMM: 128x128x32 block tile, 8 waves, 32x64 wave tile (2x4 WMMA frags),
// double-buffered LDS, fragment-order B layout, L2 prefetch.
// ---------------------------------------------------------------------------

typedef __attribute__((ext_vector_type(16))) __bf16 v16bf;
typedef __attribute__((ext_vector_type(8)))  __bf16 v8bf;
typedef __attribute__((ext_vector_type(8)))  float  v8f;

#define BM 128
#define BN 128
#define BK 32
#define LDSS 40   // LDS row stride in bf16 (32 data + 8 pad, keeps 16B align)

__device__ __forceinline__ float gelu_erf(float v) {
    // exact erf-gelu, matching jax.nn.gelu(approximate=False)
    return 0.5f * v * (1.0f + erff(v * 0.70710678118654752f));
}

// ---------------------------------------------------------------------------
// SSM scan: one wave32 per (b,h) sequence.  N=64 complex modes -> 2 per lane.
// S_n[t] = m_n * S_n[t-1] + u[t];  conv[t] = sum_n w_{h,n} * Re(S_n[t])
// ---------------------------------------------------------------------------
__global__ __launch_bounds__(32)
void ssm_scan_kernel(const float* __restrict__ x,        // [B,L,H]
                     const float* __restrict__ in_proj,  // [H,NM]
                     const float* __restrict__ out_proj, // [NM,H]
                     const float* __restrict__ freqs,    // [NM]
                     const float* __restrict__ decays,   // [NM]
                     __bf16* __restrict__ conv,          // [B*L,H] bf16
                     int L, int H, int NM) {
    const int b    = blockIdx.x / H;
    const int h    = blockIdx.x % H;
    const int lane = threadIdx.x;
    const int n0   = lane;
    const int n1   = lane + 32;     // NM == 64

    const float TWO_PI = 6.28318530717958647692f;

    // per-step complex multiplier m = exp(-exp(dec)) * e^{i*2*pi*f}
    float r0  = __expf(-__expf(decays[n0]));
    float th0 = TWO_PI * freqs[n0];
    float mr0 = r0 * __cosf(th0), mi0 = r0 * __sinf(th0);
    float w0  = in_proj[(size_t)h * NM + n0] * out_proj[(size_t)n0 * H + h];

    float r1  = __expf(-__expf(decays[n1]));
    float th1 = TWO_PI * freqs[n1];
    float mr1 = r1 * __cosf(th1), mi1 = r1 * __sinf(th1);
    float w1  = in_proj[(size_t)h * NM + n1] * out_proj[(size_t)n1 * H + h];

    float sr0 = 0.f, si0 = 0.f, sr1 = 0.f, si1 = 0.f;

    const float* xb = x    + ((size_t)b * L) * H + h;
    __bf16*      cb = conv + ((size_t)b * L) * H + h;

    for (int t0 = 0; t0 < L; t0 += 32) {
        // stage 32 inputs (one per lane), broadcast each step via shfl
        float ustage = xb[(size_t)(t0 + lane) * H];
        #pragma unroll 8
        for (int s = 0; s < 32; ++s) {
            float u = __shfl(ustage, s, 32);
            float nr0 = fmaf(mr0, sr0, fmaf(-mi0, si0, u));
            float ni0 = fmaf(mr0, si0, mi0 * sr0);
            float nr1 = fmaf(mr1, sr1, fmaf(-mi1, si1, u));
            float ni1 = fmaf(mr1, si1, mi1 * sr1);
            sr0 = nr0; si0 = ni0; sr1 = nr1; si1 = ni1;
            float p = fmaf(w0, sr0, w1 * sr1);
            // wave32 tree reduction
            p += __shfl_xor(p, 16, 32);
            p += __shfl_xor(p,  8, 32);
            p += __shfl_xor(p,  4, 32);
            p += __shfl_xor(p,  2, 32);
            p += __shfl_xor(p,  1, 32);
            if (lane == 0) cb[(size_t)(t0 + s) * H] = (__bf16)p;
        }
    }
}

// ---------------------------------------------------------------------------
// Tiled WMMA GEMM:  out[M,N] = act(A[M,K] @ W[N,K]^T + bias) (* mul)
//  - As[row][k], Bs[col][k]: both in fragment order -> contiguous LDS writes
//    during staging AND two aligned v8bf LDS loads per fragment.
//  - Double-buffered LDS; next tile's global loads issued before the WMMAs.
// ---------------------------------------------------------------------------
template <typename AT, typename OT, bool GELU, bool MUL>
__global__ __launch_bounds__(256)
void gemm_bias_act(const AT* __restrict__ A,       // [M,K]
                   const float* __restrict__ W,    // [N,K] row-major
                   const float* __restrict__ bias, // [N]
                   const __bf16* __restrict__ mul, // [M,N] or nullptr
                   OT* __restrict__ out,           // [M,N]
                   int M, int N, int K) {
    __shared__ __bf16 As[2][BM * LDSS];
    __shared__ __bf16 Bs[2][BN * LDSS];

    const int tid  = threadIdx.x;
    const int wave = tid >> 5;
    const int lane = tid & 31;
    const int wm   = wave >> 1;        // 4 waves along M (32 rows each)
    const int wn   = wave & 1;         // 2 waves along N (64 cols each)
    const int half = lane >> 4;        // lane group (0: lanes 0-15, 1: 16-31)
    const int lr   = lane & 15;

    const int m0  = blockIdx.y * BM;
    const int n0b = blockIdx.x * BN;

    // staging: 2 threads per row, 16 consecutive elements each (coalesced)
    const int sRow = tid >> 1;
    const int sCol = (tid & 1) * 16;
    const AT*    aSrc = A + (size_t)(m0 + sRow) * K + sCol;
    const float* bSrc = W + (size_t)(n0b + sRow) * K + sCol;

    AT    aReg[16];
    float bReg[16];

    // ---- preload tile 0 into buffer 0
    #pragma unroll
    for (int i = 0; i < 16; ++i) aReg[i] = aSrc[i];
    #pragma unroll
    for (int i = 0; i < 16; ++i) bReg[i] = bSrc[i];
    {
        __bf16* ad = &As[0][sRow * LDSS + sCol];
        __bf16* bd = &Bs[0][sRow * LDSS + sCol];
        #pragma unroll
        for (int i = 0; i < 16; ++i) ad[i] = (__bf16)aReg[i];
        #pragma unroll
        for (int i = 0; i < 16; ++i) bd[i] = (__bf16)bReg[i];
    }
    __syncthreads();

    v8f acc[2][4] = {};

    int cur = 0;
    for (int k0 = 0; k0 < K; k0 += BK, cur ^= 1) {
        const bool more = (k0 + BK) < K;

        // ---- issue next tile's global loads (in flight during the WMMAs)
        if (more) {
            #pragma unroll
            for (int i = 0; i < 16; ++i) aReg[i] = aSrc[k0 + BK + i];
            #pragma unroll
            for (int i = 0; i < 16; ++i) bReg[i] = bSrc[k0 + BK + i];
        }
        // ---- L2 prefetch of the tile after next (global_prefetch_b8)
        if (k0 + 2 * BK < K) {
            __builtin_prefetch(aSrc + k0 + 2 * BK, 0, 1);
            __builtin_prefetch(bSrc + k0 + 2 * BK, 0, 1);
        }

        // ---- fragment loads: two aligned 16B vector LDS reads each
        const __bf16* as = &As[cur][0];
        const __bf16* bs = &Bs[cur][0];
        v16bf afr[2], bfr[4];
        #pragma unroll
        for (int mi = 0; mi < 2; ++mi) {
            const int row = wm * 32 + mi * 16 + lr;
            v8bf lo = *(const v8bf*)&as[row * LDSS + half * 8];
            v8bf hi = *(const v8bf*)&as[row * LDSS + 16 + half * 8];
            afr[mi] = __builtin_shufflevector(
                lo, hi, 0, 1, 2, 3, 4, 5, 6, 7, 8, 9, 10, 11, 12, 13, 14, 15);
        }
        #pragma unroll
        for (int ni = 0; ni < 4; ++ni) {
            const int col = wn * 64 + ni * 16 + lr;
            v8bf lo = *(const v8bf*)&bs[col * LDSS + half * 8];
            v8bf hi = *(const v8bf*)&bs[col * LDSS + 16 + half * 8];
            bfr[ni] = __builtin_shufflevector(
                lo, hi, 0, 1, 2, 3, 4, 5, 6, 7, 8, 9, 10, 11, 12, 13, 14, 15);
        }

        // ---- 8x v_wmma_f32_16x16x32_bf16 per wave per K-step
        #pragma unroll
        for (int mi = 0; mi < 2; ++mi)
            #pragma unroll
            for (int ni = 0; ni < 4; ++ni)
                acc[mi][ni] = __builtin_amdgcn_wmma_f32_16x16x32_bf16(
                    false, afr[mi], false, bfr[ni],
                    (short)0, acc[mi][ni], false, false);

        // ---- stash next tile into the other buffer
        if (more) {
            __bf16* ad = &As[cur ^ 1][sRow * LDSS + sCol];
            __bf16* bd = &Bs[cur ^ 1][sRow * LDSS + sCol];
            #pragma unroll
            for (int i = 0; i < 16; ++i) ad[i] = (__bf16)aReg[i];
            #pragma unroll
            for (int i = 0; i < 16; ++i) bd[i] = (__bf16)bReg[i];
        }
        __syncthreads();
    }

    // ---- epilogue per f32 C/D layout: VGPR j -> row half*8+j, col = lane%16
    #pragma unroll
    for (int mi = 0; mi < 2; ++mi) {
        #pragma unroll
        for (int ni = 0; ni < 4; ++ni) {
            #pragma unroll
            for (int j = 0; j < 8; ++j) {
                const int gm = m0 + wm * 32 + mi * 16 + half * 8 + j;
                const int gn = n0b + wn * 64 + ni * 16 + lr;
                float v = acc[mi][ni][j] + bias[gn];
                if (GELU) v = gelu_erf(v);
                if (MUL)  v *= (float)mul[(size_t)gm * N + gn];
                out[(size_t)gm * N + gn] = (OT)v;
            }
        }
    }
}

// ---------------------------------------------------------------------------
extern "C" void kernel_launch(void* const* d_in, const int* in_sizes, int n_in,
                              void* d_out, int out_size, void* d_ws, size_t ws_size,
                              hipStream_t stream) {
    (void)n_in; (void)out_size; (void)ws_size;
    const float* x        = (const float*)d_in[0];
    const float* in_proj  = (const float*)d_in[1];
    const float* out_proj = (const float*)d_in[2];
    const float* freqs    = (const float*)d_in[3];
    const float* decays   = (const float*)d_in[4];
    const float* ss_W     = (const float*)d_in[5];
    const float* ss_b     = (const float*)d_in[6];
    const float* gate_W   = (const float*)d_in[7];
    const float* gate_b   = (const float*)d_in[8];
    const float* fin_W    = (const float*)d_in[9];
    const float* fin_b    = (const float*)d_in[10];

    const int NM = in_sizes[3];            // 64 modes
    const int H  = in_sizes[1] / NM;       // 1024
    const int M  = in_sizes[0] / H;        // B*L = 8192 tokens
    const int L  = 2048;                   // sequence length (problem constant)
    const int B  = M / L;
    const int N  = H, K = H;

    // workspace (bf16 intermediates): conv | ss_mid ; G reuses conv region
    __bf16* conv   = (__bf16*)d_ws;
    __bf16* ss_mid = conv + (size_t)M * H;
    __bf16* G      = conv;                 // conv dead after GEMM #1

    // 1) SSM recurrence (replaces FFT conv): 4096 independent wave32 scans
    ssm_scan_kernel<<<dim3(B * H), dim3(32), 0, stream>>>(
        x, in_proj, out_proj, freqs, decays, conv, L, H, NM);

    const dim3 grid(N / BN, M / BM), blk(256);

    // 2) ss_mid = gelu(conv @ ss_W^T + ss_b)
    gemm_bias_act<__bf16, __bf16, true, false><<<grid, blk, 0, stream>>>(
        conv, ss_W, ss_b, nullptr, ss_mid, M, N, K);

    // 3) G = gelu(x @ gate_W^T + gate_b) * ss_mid   (fused product)
    gemm_bias_act<float, __bf16, true, true><<<grid, blk, 0, stream>>>(
        x, gate_W, gate_b, ss_mid, G, M, N, K);

    // 4) out = G @ fin_W^T + fin_b   (f32 output)
    gemm_bias_act<__bf16, float, false, false><<<grid, blk, 0, stream>>>(
        G, fin_W, fin_b, nullptr, (float*)d_out, M, N, K);
}
